// CRF_55920474194063
// MI455X (gfx1250) — matrix operand order repaired and checked by reading, not compiled, not run on hardware
//
#include <hip/hip_runtime.h>
#include <hip/hip_bf16.h>

typedef _Float16 v16h __attribute__((ext_vector_type(16)));
typedef _Float16 v8h  __attribute__((ext_vector_type(8)));
typedef float    v8f  __attribute__((ext_vector_type(8)));

#define B_   64
#define S_   512
#define T_   256
#define TSED 258      // (T+2)
#define CHUNKS 8
#define TILES_PER_CHUNK 4   // 4 tiles * 16 rows = 64 timesteps per chunk
#define APAD 264            // LDS A-row stride in halves (256 + 8 pad -> conflict-free b128)

// ---------------------------------------------------------------------------
// Kernel 1: EtT[j][i] = expm1(trans[i][j]) in f16, transposed so the WMMA
// B-operand per-lane loads are contiguous in K.
// ---------------------------------------------------------------------------
__global__ void crf_prep_etrans(const float* __restrict__ trans,
                                _Float16* __restrict__ EtT) {
    const int i = threadIdx.x;   // K index (source row of trans)
    const int j = blockIdx.x;    // N index (source col of trans)
    EtT[j * T_ + i] = (_Float16)expm1f(trans[i * T_ + j]);
}

// ---------------------------------------------------------------------------
// Kernel 2: per (b, t-chunk): sum_t log( rowsum_t + exp(em_t) @ expm1(trans) )
// 16 waves/WG, each wave owns 16 output columns; B operand register-resident.
// ---------------------------------------------------------------------------
__global__ __launch_bounds__(512) void crf_forward_wmma(
    const float* __restrict__ em, const _Float16* __restrict__ EtT,
    float* __restrict__ part)
{
    __shared__ __align__(16) _Float16 Ae[16][APAD];
    __shared__ float rowsum[16];

    const int b     = blockIdx.x;
    const int chunk = blockIdx.y;
    const int wv    = threadIdx.x >> 5;   // wave id 0..15 (also staging row)
    const int lane  = threadIdx.x & 31;
    const int l15   = lane & 15;
    const int hi    = lane >> 4;          // lane half
    const int col   = wv * 16 + l15;      // output column j owned by this lane

    // ---- B operand (K=256 x N=16) once into registers.
    // B layout (contiguous K, same as A): lanes 0-15 hold K=k0..k0+15, lanes 16-31 hold K=k0+16..k0+31,
    // 2 halves per VGPR -> per lane: 16 contiguous halves of EtT[col][...]
    v16h breg[8];
    #pragma unroll
    for (int kc = 0; kc < 8; ++kc) {
        const _Float16* p = EtT + (size_t)col * T_ + kc * 32 + hi * 16;
        union { v16h v; v8h h[2]; } u;
        u.h[0] = *(const v8h*)(p);
        u.h[1] = *(const v8h*)(p + 8);
        breg[kc] = u.v;
    }

    float acc[8];
    #pragma unroll
    for (int r = 0; r < 8; ++r) acc[r] = 0.0f;

    const size_t em_b = (size_t)b * S_ * T_;

    for (int tile = 0; tile < TILES_PER_CHUNK; ++tile) {
        const int t0 = (chunk * TILES_PER_CHUNK + tile) * 16;
        __syncthreads();   // previous tile fully consumed before re-staging

        // ---- stage exp(em[b, t0+wv, :]) : wave wv handles one row, 8 f32/lane
        {
            const float* e = em + em_b + (size_t)(t0 + wv) * T_ + lane * 8;
            float4 f0 = *(const float4*)(e);
            float4 f1 = *(const float4*)(e + 4);
            float x0 = __expf(f0.x), x1 = __expf(f0.y), x2 = __expf(f0.z), x3 = __expf(f0.w);
            float x4 = __expf(f1.x), x5 = __expf(f1.y), x6 = __expf(f1.z), x7 = __expf(f1.w);
            float ps = ((x0 + x1) + (x2 + x3)) + ((x4 + x5) + (x6 + x7));
            #pragma unroll
            for (int m = 1; m < 32; m <<= 1) ps += __shfl_xor(ps, m, 32);
            if (lane == 0) rowsum[wv] = ps;
            v8h hv;
            hv[0] = (_Float16)x0; hv[1] = (_Float16)x1;
            hv[2] = (_Float16)x2; hv[3] = (_Float16)x3;
            hv[4] = (_Float16)x4; hv[5] = (_Float16)x5;
            hv[6] = (_Float16)x6; hv[7] = (_Float16)x7;
            *(v8h*)&Ae[wv][lane * 8] = hv;
        }
        __syncthreads();

        // ---- (16x256) @ (256x16) correction GEMM: 8 chained WMMAs.
        // A layout (matches B): lo lanes hold K 0..15, hi lanes hold K 16..31
        // within each 32-wide K step, contiguous per lane.
        v8f c = {};
        #pragma unroll
        for (int kc = 0; kc < 8; ++kc) {
            const _Float16* ap = &Ae[l15][kc * 32 + hi * 16];
            union { v16h v; v8h h[2]; } ua;
            ua.h[0] = *(const v8h*)(ap);
            ua.h[1] = *(const v8h*)(ap + 8);
            c = __builtin_amdgcn_wmma_f32_16x16x32_f16(false, ua.v, false, breg[kc],
                                                       (short)0, c, false, false);
        }

        // ---- epilogue: log(rowsum + delta); exclude t==0 (alpha0 uses raw em)
        #pragma unroll
        for (int r = 0; r < 8; ++r) {
            const int m = r + hi * 8;                 // local row (timestep in tile)
            const float rs = rowsum[m];
            const float lv = __logf(rs + c[r]);
            acc[r] += ((t0 + m) != 0) ? lv : 0.0f;    // cndmask, no EXEC change
        }
    }

    float tot = ((acc[0] + acc[1]) + (acc[2] + acc[3]))
              + ((acc[4] + acc[5]) + (acc[6] + acc[7]));
    tot += __shfl_xor(tot, 16, 32);                   // combine the two row-halves
    if (hi == 0) part[((size_t)chunk * B_ + b) * T_ + col] = tot;
}

// ---------------------------------------------------------------------------
// Kernel 3: per-batch logsumexp of alpha_final + gold-path score.
// ---------------------------------------------------------------------------
__global__ __launch_bounds__(256) void crf_finalize(
    const float* __restrict__ part, const float* __restrict__ em,
    const int* __restrict__ tags, const unsigned char* __restrict__ mask,
    const float* __restrict__ trans, const float* __restrict__ tse,
    float* __restrict__ out)
{
    __shared__ float redf[256];
    __shared__ int   redi[256];
    const int b = blockIdx.x;
    const int j = threadIdx.x;

    float s = 0.0f;
    #pragma unroll
    for (int c = 0; c < CHUNKS; ++c) s += part[((size_t)c * B_ + b) * T_ + j];
    const float v = em[(size_t)b * S_ * T_ + j]        // em[b,0,j]
                  + tse[T_ * TSED + j]                 // tse[start, j]
                  + s
                  + tse[j * TSED + (T_ + 1)];          // tse[j, end]

    // max-shifted logsumexp over j (alpha ~ 3e3, direct exp would overflow)
    redf[j] = v; __syncthreads();
    for (int st = 128; st > 0; st >>= 1) {
        if (j < st) redf[j] = fmaxf(redf[j], redf[j + st]);
        __syncthreads();
    }
    const float mx = redf[0]; __syncthreads();
    redf[j] = __expf(v - mx); __syncthreads();
    for (int st = 128; st > 0; st >>= 1) {
        if (j < st) redf[j] += redf[j + st];
        __syncthreads();
    }
    const float fwd = mx + __logf(redf[0]);
    __syncthreads();

    // gold-path score: emissions gather + transition gather + mask count
    float sc = 0.0f; int cnt = 0;
    #pragma unroll
    for (int t = j; t < S_; t += 256) {
        const int base = b * S_ + t;
        const int mk = mask[base] ? 1 : 0;
        cnt += mk;
        const float mf = (float)mk;
        sc += em[(size_t)b * S_ * T_ + (size_t)t * T_ + tags[base]] * mf;
        if (t >= 1) sc += trans[tags[base] * T_ + tags[base - 1]] * mf;
    }
    redf[j] = sc; redi[j] = cnt; __syncthreads();
    for (int st = 128; st > 0; st >>= 1) {
        if (j < st) { redf[j] += redf[j + st]; redi[j] += redi[j + st]; }
        __syncthreads();
    }
    if (j == 0) {
        float score = redf[0];
        const int last_idx = redi[0] - 1;
        score += tse[T_ * TSED + tags[b * S_]];               // tse[start, tag0]
        score += tse[tags[b * S_ + last_idx] * TSED + (T_ + 1)]; // tse[lastTag, end]
        out[b] = fwd - score;
    }
}

// ---------------------------------------------------------------------------
extern "C" void kernel_launch(void* const* d_in, const int* in_sizes, int n_in,
                              void* d_out, int out_size, void* d_ws, size_t ws_size,
                              hipStream_t stream) {
    (void)in_sizes; (void)n_in; (void)out_size; (void)ws_size;
    const float*         em    = (const float*)d_in[0];
    const int*           tags  = (const int*)d_in[1];
    const unsigned char* mask  = (const unsigned char*)d_in[2];
    const float*         trans = (const float*)d_in[3];
    const float*         tse   = (const float*)d_in[4];
    float* out = (float*)d_out;

    _Float16* EtT = (_Float16*)d_ws;                                   // 128 KB
    float* part   = (float*)((char*)d_ws + (size_t)T_ * T_ * sizeof(_Float16)); // 512 KB

    crf_prep_etrans<<<T_, T_, 0, stream>>>(trans, EtT);
    crf_forward_wmma<<<dim3(B_, CHUNKS), 512, 0, stream>>>(em, EtT, part);
    crf_finalize<<<B_, 256, 0, stream>>>(part, em, tags, mask, trans, tse, out);
}
